// RWKVv7Separator_25769803895
// MI455X (gfx1250) — compile-verified, matrix operand order checked by reading, not run on hardware
//
#include <hip/hip_runtime.h>
#include <hip/hip_bf16.h>
#include <cstddef>

typedef __bf16 bf16_t;
typedef __attribute__((ext_vector_type(16))) __bf16 v16bf;
typedef __attribute__((ext_vector_type(8)))  float  v8f;

#define ACT_NONE    0
#define ACT_RELU2   1
#define ACT_SIGMOID 2
#define ACT_TANH    3
#define ACT_GELU    4

#if defined(__AMDGCN__) && __has_builtin(__builtin_amdgcn_global_load_async_to_lds_b128) && __has_builtin(__builtin_amdgcn_s_wait_asynccnt)
#define HAVE_ASYNC_LDS 1
#else
#define HAVE_ASYNC_LDS 0
#endif

#if HAVE_ASYNC_LDS
// builtin expects pointers to 16-byte int vectors; AS(1) == __device__, AS(3) == LDS.
// generic -> as(1): numerically identical; generic -> as(3): low 32 bits are the LDS offset
typedef int v4i_b128 __attribute__((vector_size(16)));
#define TO_GLOBAL(p) ((__attribute__((address_space(1))) v4i_b128*)(unsigned long long)(p))
#define TO_LDS(p)    ((__attribute__((address_space(3))) v4i_b128*)(unsigned int)(unsigned long long)(p))
#endif

namespace {
constexpr int Bq = 2, Tq = 1024, Cc = 512, Hh = 256, NHc = 4, NLc = 6, DLc = 64, HHc = 512;
constexpr int Mrows = Bq * Tq;                 // 2048 token rows
constexpr size_t SLOT = (size_t)Mrows * 1024;  // floats per workspace slot (8 MB)
constexpr float LNX_EPS = 1e-5f * 64.0f * 64.0f;
}

__device__ __forceinline__ float apply_act(float x, int act) {
  switch (act) {
    case ACT_RELU2:   { float r = fmaxf(x, 0.f); return r * r; }
    case ACT_SIGMOID: return 1.f / (1.f + __expf(-x));
    case ACT_TANH:    return tanhf(x);
    case ACT_GELU:    return 0.5f * x * (1.f + erff(x * 0.70710678118f));
    default:          return x;
  }
}

// ---------------------------------------------------------------------------
// Weight pre-pass: W[K,N] f32 -> Wt[N,K] bf16 (transposed so the GEMM's B-tile
// is K-contiguous per output column -> vector LDS traffic + async staging).
// ---------------------------------------------------------------------------
__global__ __launch_bounds__(256)
void transpose_cvt_k(const float* __restrict__ W, bf16_t* __restrict__ Wt,
                     int K, int N)
{
  __shared__ float tile[32][33];
  const int kb = blockIdx.x * 32, nb = blockIdx.y * 32;
  const int tx = threadIdx.x & 31, ty = threadIdx.x >> 5;   // 32 x 8
#pragma unroll
  for (int i = ty; i < 32; i += 8)
    tile[i][tx] = W[(size_t)(kb + i) * N + nb + tx];
  __syncthreads();
#pragma unroll
  for (int i = ty; i < 32; i += 8)
    Wt[(size_t)(nb + i) * K + kb + tx] = (bf16_t)tile[tx][i];
}

// ---------------------------------------------------------------------------
// WMMA GEMM: C[M,N] = act(A[M,K] @ W[K,N] + bias), weights given as bf16
// Wt[N,K]. 128 threads = 4 waves; each wave owns a 32x32 quadrant of the
// 64x64 block tile = 2x2 v_wmma_f32_16x16x32_bf16 fragments.
// Ping-pong double-buffered LDS: tile k+1 is staged (async B-loads via
// ASYNCcnt + bf16-converted A stores) while tile k runs on the WMMA pipe;
// one s_wait_asynccnt + barrier per iteration.
// ---------------------------------------------------------------------------
__global__ __launch_bounds__(128)
void gemm_bf16_wmma(const float* __restrict__ A, const bf16_t* __restrict__ Wt,
                    const float* __restrict__ bias, float* __restrict__ C,
                    int M, int N, int K, int act)
{
  __shared__ bf16_t sA[2][64][40];   // [buf][m][k] 64x32 (+pad, rows 16B-aligned)
  __shared__ bf16_t sB[2][64][40];   // [buf][n][k]

  const int tid  = threadIdx.x;
  const int wave = tid >> 5, lane = tid & 31;
  const int m0 = blockIdx.y * 64, n0 = blockIdx.x * 64;
  const int wm = (wave >> 1) * 32, wn = (wave & 1) * 32;
  const int l15 = lane & 15;
  const int ks  = (lane & 16) ? 8 : 0;    // ISA 16-bit A/B-frag K split
  const int roff = (lane & 16) ? 8 : 0;   // ISA 32-bit C/D row split
  const int sr = tid >> 1, sc = (tid & 1) * 16;   // staging row/col

  auto stageB = [&](int p, int k0) {
    const bf16_t* src = Wt + (size_t)(n0 + sr) * K + k0 + sc;
#if HAVE_ASYNC_LDS
    __builtin_amdgcn_global_load_async_to_lds_b128(TO_GLOBAL(src),
                                                   TO_LDS(&sB[p][sr][sc]), 0, 0);
    __builtin_amdgcn_global_load_async_to_lds_b128(TO_GLOBAL(src + 8),
                                                   TO_LDS(&sB[p][sr][sc + 8]), 0, 0);
#else
    const uint4* s4 = (const uint4*)src;
    *(uint4*)&sB[p][sr][sc]     = s4[0];
    *(uint4*)&sB[p][sr][sc + 8] = s4[1];
#endif
  };
  auto stageA = [&](int p, int k0) {
    const float* src = A + (size_t)(m0 + sr) * K + k0 + sc;
    if (k0 + 32 < K) __builtin_prefetch(src + 32, 0, 1);
#pragma unroll
    for (int i = 0; i < 16; i += 4) {
      float4 f = *(const float4*)(src + i);
      sA[p][sr][sc + i + 0] = (bf16_t)f.x;
      sA[p][sr][sc + i + 1] = (bf16_t)f.y;
      sA[p][sr][sc + i + 2] = (bf16_t)f.z;
      sA[p][sr][sc + i + 3] = (bf16_t)f.w;
    }
  };

  v8f acc[2][2] = {};

  // prologue: stage tile 0 into buffer 0
  stageB(0, 0);
  stageA(0, 0);
#if HAVE_ASYNC_LDS
  __builtin_amdgcn_s_wait_asynccnt(0);
#endif
  __syncthreads();

  const int nk = K >> 5;
  for (int kt = 0; kt < nk; ++kt) {
    const int p = kt & 1;
    if (kt + 1 < nk) {            // stage next tile into the other buffer
      stageB(p ^ 1, (kt + 1) * 32);
      stageA(p ^ 1, (kt + 1) * 32);
    }

    // fragment loads: contiguous 8-halfword chunks -> ds_load_b128
    v16bf af[2], bv[2];
#pragma unroll
    for (int q = 0; q < 2; ++q) {
      int ra = wm + 16 * q + l15;
      int rb = wn + 16 * q + l15;
#pragma unroll
      for (int i = 0; i < 8; ++i) {
        af[q][i]     = sA[p][ra][ks + i];
        af[q][i + 8] = sA[p][ra][ks + 16 + i];
        bv[q][i]     = sB[p][rb][ks + i];
        bv[q][i + 8] = sB[p][rb][ks + 16 + i];
      }
    }
#pragma unroll
    for (int mi = 0; mi < 2; ++mi)
#pragma unroll
      for (int ni = 0; ni < 2; ++ni)
        acc[mi][ni] = __builtin_amdgcn_wmma_f32_16x16x32_bf16(
            false, af[mi], false, bv[ni], (short)0, acc[mi][ni], false, false);

    if (kt + 1 < nk) {
#if HAVE_ASYNC_LDS
      __builtin_amdgcn_s_wait_asynccnt(0);
#endif
      __syncthreads();
    }
  }

#pragma unroll
  for (int mi = 0; mi < 2; ++mi)
#pragma unroll
    for (int ni = 0; ni < 2; ++ni) {
      int colg = n0 + wn + 16 * ni + l15;
      float bv_ = bias ? bias[colg] : 0.f;
#pragma unroll
      for (int e = 0; e < 8; ++e) {
        int rowg = m0 + wm + 16 * mi + e + roff;
        C[(size_t)rowg * N + colg] = apply_act(acc[mi][ni][e] + bv_, act);
      }
    }
}

// ---------------------------------------------------------------------------
// WKV7 scan: one 64-thread block per (batch, head). Thread i owns state row
// S[i][:] (64 f32 in VGPRs). Per step: broadcast r,d,k,v,a,b via LDS, then
// sa_i = S[i]·a ; S[i][j] = S[i][j]*d[j] + sa_i*b[j] + v_i*k[j] ; y_i = S[i]·r
// ---------------------------------------------------------------------------
__global__ __launch_bounds__(64)
void wkv7_scan(const float* __restrict__ r, const float* __restrict__ d,
               const float* __restrict__ k, const float* __restrict__ v,
               const float* __restrict__ aa, const float* __restrict__ bb,
               float* __restrict__ y, int T, int dir)
{
  const int b = blockIdx.x / NHc;
  const int h = blockIdx.x % NHc;
  const int i = threadIdx.x;
  __shared__ float sr[64], sd[64], sk[64], sv[64], sa[64], sb[64];

  float S[64];
#pragma unroll
  for (int j = 0; j < 64; ++j) S[j] = 0.f;

  for (int tt = 0; tt < T; ++tt) {
    int t = dir ? (T - 1 - tt) : tt;
    size_t base = ((size_t)b * T + t) * Hh + h * 64 + i;
    sr[i] = r[base]; sd[i] = d[base]; sk[i] = k[base];
    sv[i] = v[base]; sa[i] = aa[base]; sb[i] = bb[base];
    __syncthreads();

    float sav = 0.f;
#pragma unroll 8
    for (int j = 0; j < 64; ++j) sav += S[j] * sa[j];
    float vi = sv[i], yv = 0.f;
#pragma unroll 8
    for (int j = 0; j < 64; ++j) {
      float s = S[j] * sd[j] + sav * sb[j] + vi * sk[j];
      S[j] = s;
      yv += s * sr[j];
    }
    y[base] = yv;
    __syncthreads();
  }
}

// ---------------------------------------------------------------------------
// LayerNorm over last dim (C channels per row), block = 256 threads per row
// ---------------------------------------------------------------------------
__global__ __launch_bounds__(256)
void layernorm_k(const float* __restrict__ x, const float* __restrict__ g,
                 const float* __restrict__ b, float* __restrict__ o, int C)
{
  __shared__ float red[256];
  const int row = blockIdx.x, tid = threadIdx.x;
  const float* xr = x + (size_t)row * C;
  float s = 0.f;
  for (int c = tid; c < C; c += 256) s += xr[c];
  red[tid] = s; __syncthreads();
  for (int st = 128; st > 0; st >>= 1) { if (tid < st) red[tid] += red[tid + st]; __syncthreads(); }
  float mu = red[0] / C; __syncthreads();
  float v = 0.f;
  for (int c = tid; c < C; c += 256) { float dd = xr[c] - mu; v += dd * dd; }
  red[tid] = v; __syncthreads();
  for (int st = 128; st > 0; st >>= 1) { if (tid < st) red[tid] += red[tid + st]; __syncthreads(); }
  float inv = rsqrtf(red[0] / C + 1e-5f);
  for (int c = tid; c < C; c += 256)
    o[(size_t)row * C + c] = (xr[c] - mu) * inv * g[c] + b[c];
}

// GroupNorm: grid = rows*NH, 64 threads per group; scale/shift by full-channel g,b
__global__ __launch_bounds__(64)
void groupnorm_k(const float* __restrict__ x, const float* __restrict__ g,
                 const float* __restrict__ b, float* __restrict__ o)
{
  __shared__ float red[64];
  const int row = blockIdx.x / NHc, h = blockIdx.x % NHc, i = threadIdx.x;
  const int col = h * 64 + i;
  size_t base = (size_t)row * Hh + col;
  float xv = x[base];
  red[i] = xv; __syncthreads();
  for (int st = 32; st > 0; st >>= 1) { if (i < st) red[i] += red[i + st]; __syncthreads(); }
  float mu = red[0] * (1.f / 64.f); __syncthreads();
  float dd = xv - mu;
  red[i] = dd * dd; __syncthreads();
  for (int st = 32; st > 0; st >>= 1) { if (i < st) red[i] += red[i + st]; __syncthreads(); }
  float inv = rsqrtf(red[0] * (1.f / 64.f) + LNX_EPS);
  o[base] = dd * inv * g[col] + b[col];
}

// token shift-mix: o = x + (shift(x)-x)*mu ; dir flips which neighbor is "previous"
__global__ __launch_bounds__(256)
void mix_shift_k(const float* __restrict__ x, const float* __restrict__ mu,
                 float* __restrict__ o, int dir)
{
  int idx = blockIdx.x * 256 + threadIdx.x;
  int c = idx % Hh, t = (idx / Hh) % Tq;
  float xs = 0.f;
  if (!dir) { if (t > 0)      xs = x[idx - Hh]; }
  else      { if (t < Tq - 1) xs = x[idx + Hh]; }
  float xv = x[idx];
  o[idx] = xv + (xs - xv) * mu[c];
}

// w = -softplus(-(w0 + z)) - 0.5   (in place on z)
__global__ __launch_bounds__(256)
void w_epilogue_k(const float* __restrict__ w0, float* __restrict__ w)
{
  int idx = blockIdx.x * 256 + threadIdx.x;
  float u = w0[idx % Hh] + w[idx];
  w[idx] = -log1pf(__expf(-u)) - 0.5f;
}

__global__ __launch_bounds__(256)
void sigmoid_bias_k(const float* __restrict__ bias, const float* __restrict__ z,
                    float* __restrict__ o)
{
  int idx = blockIdx.x * 256 + threadIdx.x;
  o[idx] = 1.f / (1.f + __expf(-(bias[idx % Hh] + z[idx])));
}

__global__ __launch_bounds__(256)
void value_mix_k(float* __restrict__ v, const float* __restrict__ vf,
                 const float* __restrict__ s)
{
  int idx = blockIdx.x * 256 + threadIdx.x;
  float vv = v[idx];
  v[idx] = vv + (vf[idx] - vv) * s[idx];
}

// per-(row,head): kk = normalize(k*k_k); aa=-kk; bb=kk*a; k*=(1+(a-1)*k_a); d=exp(-exp(w))
__global__ __launch_bounds__(64)
void wkv_prep_k(float* __restrict__ k, const float* __restrict__ a,
                const float* __restrict__ w, const float* __restrict__ k_k,
                const float* __restrict__ k_a, float* __restrict__ d,
                float* __restrict__ aa, float* __restrict__ bb)
{
  __shared__ float red[64];
  const int row = blockIdx.x / NHc, h = blockIdx.x % NHc, i = threadIdx.x;
  const int col = h * 64 + i;
  size_t base = (size_t)row * Hh + col;
  float kv = k[base], av = a[base];
  float kk = kv * k_k[col];
  red[i] = kk * kk; __syncthreads();
  for (int st = 32; st > 0; st >>= 1) { if (i < st) red[i] += red[i + st]; __syncthreads(); }
  float nrm = sqrtf(red[0]);
  float kkn = kk / fmaxf(nrm, 1e-12f);
  aa[base] = -kkn;
  bb[base] = kkn * av;
  k[base] = kv * (1.f + (av - 1.f) * k_a[col]);
  d[base] = __expf(-__expf(w[base]));
}

// per-(row,head): dot = sum(r*k*r_k); o = (ygn + dot*v) * g
__global__ __launch_bounds__(64)
void bonus_gate_k(const float* __restrict__ ygn, const float* __restrict__ r,
                  const float* __restrict__ k, const float* __restrict__ rk,
                  const float* __restrict__ v, const float* __restrict__ g,
                  float* __restrict__ o)
{
  __shared__ float red[64];
  const int row = blockIdx.x / NHc, h = blockIdx.x % NHc, i = threadIdx.x;
  const int col = h * 64 + i;
  size_t base = (size_t)row * Hh + col;
  red[i] = r[base] * k[base] * rk[col]; __syncthreads();
  for (int st = 32; st > 0; st >>= 1) { if (i < st) red[i] += red[i + st]; __syncthreads(); }
  float dot = red[0];
  o[base] = (ygn[base] + dot * v[base]) * g[base];
}

__global__ __launch_bounds__(256)
void add_inplace_k(float* __restrict__ x, const float* __restrict__ h)
{
  int idx = blockIdx.x * 256 + threadIdx.x;
  x[idx] += h[idx];
}

__global__ __launch_bounds__(256)
void avg2_k(const float* __restrict__ a, const float* __restrict__ b, float* __restrict__ o)
{
  int idx = blockIdx.x * 256 + threadIdx.x;
  o[idx] = 0.5f * (a[idx] + b[idx]);
}

__global__ __launch_bounds__(256)
void combine_k(const float* __restrict__ z, const float* __restrict__ m,
               const float* __restrict__ r, float* __restrict__ y)
{
  int idx = blockIdx.x * 256 + threadIdx.x;
  y[idx] = m[idx] * z[idx] + r[idx];
}

// ===========================================================================
// Host orchestration
// ===========================================================================
namespace {

struct TmixP {
  const float *Wk, *Wo, *Wr, *Wv, *a0, *a1, *a2, *g1, *g2, *k_a, *k_k,
              *lnx_b, *lnx_g, *r_k, *v0, *v1, *v2, *w0, *w1, *w2,
              *x_a, *x_g, *x_k, *x_r, *x_v, *x_w;
};
struct LayerPm {
  const float *cWk, *cWv, *cx_k, *ln1_b, *ln1_g, *ln2_b, *ln2_g;
  TmixP t;
};
struct LayerW {   // pre-transposed bf16 weights
  const bf16_t *Wr, *Wk, *Wv, *Wo, *w1, *w2, *a1, *a2, *g1, *g2,
               *v1, *v2, *cWk, *cWv;
};
struct RHeadP { const float *b1, *b2, *ln_b, *ln_g, *w1, *w2; };
struct MHeadP { const float *b, *ln_b, *ln_g, *w; };

inline const float* nxt(void* const* d_in, int n_in, int& i) {
  int j = (i < n_in) ? i : (n_in > 0 ? n_in - 1 : 0);
  ++i;
  return (const float*)d_in[j];
}

// sorted pytree order within a layer: cmix{Wk,Wv,x_k}, ln1_b, ln1_g, ln2_b,
// ln2_g, tmix{Wk,Wo,Wr,Wv,a0,a1,a2,g1,g2,k_a,k_k,lnx_b,lnx_g,r_k,[v0,v1,v2],
//             w0,w1,w2,x_a,x_g,x_k,x_r,x_v,x_w}
void read_layer(LayerPm& L, void* const* d_in, int n_in, int& i, int layer_id) {
  L.cWk = nxt(d_in, n_in, i); L.cWv = nxt(d_in, n_in, i); L.cx_k = nxt(d_in, n_in, i);
  L.ln1_b = nxt(d_in, n_in, i); L.ln1_g = nxt(d_in, n_in, i);
  L.ln2_b = nxt(d_in, n_in, i); L.ln2_g = nxt(d_in, n_in, i);
  TmixP& t = L.t;
  t.Wk = nxt(d_in, n_in, i); t.Wo = nxt(d_in, n_in, i);
  t.Wr = nxt(d_in, n_in, i); t.Wv = nxt(d_in, n_in, i);
  t.a0 = nxt(d_in, n_in, i); t.a1 = nxt(d_in, n_in, i); t.a2 = nxt(d_in, n_in, i);
  t.g1 = nxt(d_in, n_in, i); t.g2 = nxt(d_in, n_in, i);
  t.k_a = nxt(d_in, n_in, i); t.k_k = nxt(d_in, n_in, i);
  t.lnx_b = nxt(d_in, n_in, i); t.lnx_g = nxt(d_in, n_in, i);
  t.r_k = nxt(d_in, n_in, i);
  if (layer_id > 0) { t.v0 = nxt(d_in, n_in, i); t.v1 = nxt(d_in, n_in, i); t.v2 = nxt(d_in, n_in, i); }
  else { t.v0 = t.v1 = t.v2 = nullptr; }
  t.w0 = nxt(d_in, n_in, i); t.w1 = nxt(d_in, n_in, i); t.w2 = nxt(d_in, n_in, i);
  t.x_a = nxt(d_in, n_in, i); t.x_g = nxt(d_in, n_in, i); t.x_k = nxt(d_in, n_in, i);
  t.x_r = nxt(d_in, n_in, i); t.x_v = nxt(d_in, n_in, i); t.x_w = nxt(d_in, n_in, i);
}

// bump-allocating weight converter
struct Cvt {
  bf16_t* arena; size_t off; hipStream_t s;
  const bf16_t* operator()(const float* W, int K, int N) {
    bf16_t* dst = arena + off;
    off = (off + (size_t)K * N + 127) & ~(size_t)127;
    dim3 g(K / 32, N / 32), b(256);
    transpose_cvt_k<<<g, b, 0, s>>>(W, dst, K, N);
    return dst;
  }
};

void cvt_layer(LayerW& Wl, const LayerPm& L, int layer_id, Cvt& cv) {
  Wl.Wr = cv(L.t.Wr, Hh, Hh);  Wl.Wk = cv(L.t.Wk, Hh, Hh);
  Wl.Wv = cv(L.t.Wv, Hh, Hh);  Wl.Wo = cv(L.t.Wo, Hh, Hh);
  Wl.w1 = cv(L.t.w1, Hh, DLc); Wl.w2 = cv(L.t.w2, DLc, Hh);
  Wl.a1 = cv(L.t.a1, Hh, DLc); Wl.a2 = cv(L.t.a2, DLc, Hh);
  Wl.g1 = cv(L.t.g1, Hh, DLc); Wl.g2 = cv(L.t.g2, DLc, Hh);
  if (layer_id > 0) { Wl.v1 = cv(L.t.v1, Hh, DLc); Wl.v2 = cv(L.t.v2, DLc, Hh); }
  else { Wl.v1 = Wl.v2 = nullptr; }
  Wl.cWk = cv(L.cWk, Hh, 4 * Hh); Wl.cWv = cv(L.cWv, 4 * Hh, Hh);
}

inline void gemm(const float* A, const bf16_t* Wt, const float* bias, float* C,
                 int M, int N, int K, int act, hipStream_t s) {
  dim3 g(N / 64, M / 64), b(128);
  gemm_bf16_wmma<<<g, b, 0, s>>>(A, Wt, bias, C, M, N, K, act);
}

inline int ewH_blocks() { return (Mrows * Hh) / 256; }

struct Bufs {
  float *xn1, *xn2, *xm, *xv, *r, *w, *k, *v, *a, *g, *t1, *t2,
        *aa, *bb, *d, *y, *ygn, *h, *big, *vfirst;
};

void run_layer(const LayerPm& L, const LayerW& Wl, float* x, int layer_id,
               int dir, const Bufs& B_, hipStream_t s) {
  const int nb = ewH_blocks();
  layernorm_k<<<Mrows, 256, 0, s>>>(x, L.ln1_g, L.ln1_b, B_.xn1, Hh);
  layernorm_k<<<Mrows, 256, 0, s>>>(x, L.ln2_g, L.ln2_b, B_.xn2, Hh);  // pre-residual

  // ---- tmix ----
  mix_shift_k<<<nb, 256, 0, s>>>(B_.xn1, L.t.x_r, B_.xm, dir);
  gemm(B_.xm, Wl.Wr, nullptr, B_.r, Mrows, Hh, Hh, ACT_NONE, s);

  mix_shift_k<<<nb, 256, 0, s>>>(B_.xn1, L.t.x_w, B_.xm, dir);
  gemm(B_.xm, Wl.w1, nullptr, B_.t1, Mrows, DLc, Hh, ACT_TANH, s);
  gemm(B_.t1, Wl.w2, nullptr, B_.w, Mrows, Hh, DLc, ACT_NONE, s);
  w_epilogue_k<<<nb, 256, 0, s>>>(L.t.w0, B_.w);

  mix_shift_k<<<nb, 256, 0, s>>>(B_.xn1, L.t.x_k, B_.xm, dir);
  gemm(B_.xm, Wl.Wk, nullptr, B_.k, Mrows, Hh, Hh, ACT_NONE, s);

  mix_shift_k<<<nb, 256, 0, s>>>(B_.xn1, L.t.x_v, B_.xv, dir);
  gemm(B_.xv, Wl.Wv, nullptr, B_.v, Mrows, Hh, Hh, ACT_NONE, s);
  if (layer_id == 0) {
    (void)hipMemcpyAsync(B_.vfirst, B_.v, (size_t)Mrows * Hh * sizeof(float),
                         hipMemcpyDeviceToDevice, s);
  } else {
    gemm(B_.xv, Wl.v1, nullptr, B_.t1, Mrows, DLc, Hh, ACT_NONE, s);
    gemm(B_.t1, Wl.v2, nullptr, B_.t2, Mrows, Hh, DLc, ACT_NONE, s);
    sigmoid_bias_k<<<nb, 256, 0, s>>>(L.t.v0, B_.t2, B_.t2);
    value_mix_k<<<nb, 256, 0, s>>>(B_.v, B_.vfirst, B_.t2);
  }

  mix_shift_k<<<nb, 256, 0, s>>>(B_.xn1, L.t.x_a, B_.xm, dir);
  gemm(B_.xm, Wl.a1, nullptr, B_.t1, Mrows, DLc, Hh, ACT_NONE, s);
  gemm(B_.t1, Wl.a2, nullptr, B_.t2, Mrows, Hh, DLc, ACT_NONE, s);
  sigmoid_bias_k<<<nb, 256, 0, s>>>(L.t.a0, B_.t2, B_.a);

  mix_shift_k<<<nb, 256, 0, s>>>(B_.xn1, L.t.x_g, B_.xm, dir);
  gemm(B_.xm, Wl.g1, nullptr, B_.t1, Mrows, DLc, Hh, ACT_SIGMOID, s);
  gemm(B_.t1, Wl.g2, nullptr, B_.g, Mrows, Hh, DLc, ACT_NONE, s);

  wkv_prep_k<<<Mrows * NHc, 64, 0, s>>>(B_.k, B_.a, B_.w, L.t.k_k, L.t.k_a,
                                        B_.d, B_.aa, B_.bb);
  wkv7_scan<<<Bq * NHc, 64, 0, s>>>(B_.r, B_.d, B_.k, B_.v, B_.aa, B_.bb,
                                    B_.y, Tq, dir);
  groupnorm_k<<<Mrows * NHc, 64, 0, s>>>(B_.y, L.t.lnx_g, L.t.lnx_b, B_.ygn);
  bonus_gate_k<<<Mrows * NHc, 64, 0, s>>>(B_.ygn, B_.r, B_.k, L.t.r_k, B_.v,
                                          B_.g, B_.y);
  gemm(B_.y, Wl.Wo, nullptr, B_.h, Mrows, Hh, Hh, ACT_NONE, s);
  add_inplace_k<<<nb, 256, 0, s>>>(x, B_.h);

  // ---- cmix ----
  mix_shift_k<<<nb, 256, 0, s>>>(B_.xn2, L.cx_k, B_.xm, dir);
  gemm(B_.xm, Wl.cWk, nullptr, B_.big, Mrows, 4 * Hh, Hh, ACT_RELU2, s);
  gemm(B_.big, Wl.cWv, nullptr, B_.h, Mrows, Hh, 4 * Hh, ACT_NONE, s);
  add_inplace_k<<<nb, 256, 0, s>>>(x, B_.h);
}

}  // namespace

extern "C" void kernel_launch(void* const* d_in, const int* in_sizes, int n_in,
                              void* d_out, int out_size, void* d_ws, size_t ws_size,
                              hipStream_t stream) {
  (void)in_sizes; (void)out_size; (void)ws_size;
  int pi = 0;
  const float* z_mix = nxt(d_in, n_in, pi);  // (B,T,C)

  // params pytree, sorted keys: bwd[0..5], down_b, down_w, fwd[0..5],
  // head_m1, head_m2, head_r1, head_r2, up_b, up_w
  LayerPm bwd[NLc], fwd[NLc];
  for (int l = 0; l < NLc; ++l) read_layer(bwd[l], d_in, n_in, pi, l);
  const float* down_b = nxt(d_in, n_in, pi);
  const float* down_w = nxt(d_in, n_in, pi);
  for (int l = 0; l < NLc; ++l) read_layer(fwd[l], d_in, n_in, pi, l);
  MHeadP m1, m2; RHeadP r1, r2;
  m1.b = nxt(d_in, n_in, pi); m1.ln_b = nxt(d_in, n_in, pi);
  m1.ln_g = nxt(d_in, n_in, pi); m1.w = nxt(d_in, n_in, pi);
  m2.b = nxt(d_in, n_in, pi); m2.ln_b = nxt(d_in, n_in, pi);
  m2.ln_g = nxt(d_in, n_in, pi); m2.w = nxt(d_in, n_in, pi);
  r1.b1 = nxt(d_in, n_in, pi); r1.b2 = nxt(d_in, n_in, pi);
  r1.ln_b = nxt(d_in, n_in, pi); r1.ln_g = nxt(d_in, n_in, pi);
  r1.w1 = nxt(d_in, n_in, pi); r1.w2 = nxt(d_in, n_in, pi);
  r2.b1 = nxt(d_in, n_in, pi); r2.b2 = nxt(d_in, n_in, pi);
  r2.ln_b = nxt(d_in, n_in, pi); r2.ln_g = nxt(d_in, n_in, pi);
  r2.w1 = nxt(d_in, n_in, pi); r2.w2 = nxt(d_in, n_in, pi);
  const float* up_b = nxt(d_in, n_in, pi);
  const float* up_w = nxt(d_in, n_in, pi);

  // workspace slots (8 MB each)
  float* ws = (float*)d_ws;
  auto S = [&](int i) { return ws + (size_t)i * SLOT; };
  float* xf = S(0);
  float* xb = S(1);
  Bufs Bf;
  Bf.vfirst = S(2);  Bf.xn1 = S(3);  Bf.xn2 = S(4);  Bf.xm = S(5);
  Bf.xv = S(6);      Bf.r = S(7);    Bf.w = S(8);    Bf.k = S(9);
  Bf.v = S(10);      Bf.a = S(11);   Bf.g = S(12);   Bf.t1 = S(13);
  Bf.t2 = S(14);     Bf.aa = S(15);  Bf.bb = S(16);  Bf.d = S(17);
  Bf.y = S(18);      Bf.ygn = S(19); Bf.h = S(20);   Bf.big = S(21);
  float* xh  = S(22);   // averaged hidden (M,H)
  float* hC  = S(23);   // up-projected (M,C)
  float* tC  = S(24);   // head LN temp (M,C)
  float* tH  = S(25);   // head hidden (M,HH)

  // ---- weight pre-pass: transpose + convert to bf16 once per call ----
  Cvt cv{(bf16_t*)(ws + 26 * SLOT), 0, stream};
  LayerW fwdW[NLc], bwdW[NLc];
  for (int l = 0; l < NLc; ++l) cvt_layer(bwdW[l], bwd[l], l, cv);
  for (int l = 0; l < NLc; ++l) cvt_layer(fwdW[l], fwd[l], l, cv);
  const bf16_t* t_down = cv(down_w, Cc, Hh);
  const bf16_t* t_up   = cv(up_w, Hh, Cc);
  const bf16_t* t_r1w1 = cv(r1.w1, Cc, HHc); const bf16_t* t_r1w2 = cv(r1.w2, HHc, Cc);
  const bf16_t* t_r2w1 = cv(r2.w1, Cc, HHc); const bf16_t* t_r2w2 = cv(r2.w2, HHc, Cc);
  const bf16_t* t_m1w  = cv(m1.w, Cc, Cc);   const bf16_t* t_m2w  = cv(m2.w, Cc, Cc);

  const size_t MC = (size_t)Mrows * Cc;
  float* out = (float*)d_out;
  float* o_y1 = out;          float* o_y2 = out + MC;
  float* o_m1 = out + 2 * MC; float* o_m2 = out + 3 * MC;
  float* o_r1 = out + 4 * MC; float* o_r2 = out + 5 * MC;

  // down-projection, shared by both directions
  gemm(z_mix, t_down, down_b, xf, Mrows, Hh, Cc, ACT_NONE, stream);
  (void)hipMemcpyAsync(xb, xf, (size_t)Mrows * Hh * sizeof(float),
                       hipMemcpyDeviceToDevice, stream);

  for (int l = 0; l < NLc; ++l) run_layer(fwd[l], fwdW[l], xf, l, /*dir=*/0, Bf, stream);
  for (int l = 0; l < NLc; ++l) run_layer(bwd[l], bwdW[l], xb, l, /*dir=*/1, Bf, stream);

  avg2_k<<<ewH_blocks(), 256, 0, stream>>>(xf, xb, xh);
  gemm(xh, t_up, up_b, hC, Mrows, Cc, Hh, ACT_NONE, stream);

  const int nbC = (int)(MC / 256);
  // residual heads: LN -> GELU MLP
  layernorm_k<<<Mrows, 256, 0, stream>>>(hC, r1.ln_g, r1.ln_b, tC, Cc);
  gemm(tC, t_r1w1, r1.b1, tH, Mrows, HHc, Cc, ACT_GELU, stream);
  gemm(tH, t_r1w2, r1.b2, o_r1, Mrows, Cc, HHc, ACT_NONE, stream);
  layernorm_k<<<Mrows, 256, 0, stream>>>(hC, r2.ln_g, r2.ln_b, tC, Cc);
  gemm(tC, t_r2w1, r2.b1, tH, Mrows, HHc, Cc, ACT_GELU, stream);
  gemm(tH, t_r2w2, r2.b2, o_r2, Mrows, Cc, HHc, ACT_NONE, stream);
  // mask heads: LN -> sigmoid linear
  layernorm_k<<<Mrows, 256, 0, stream>>>(hC, m1.ln_g, m1.ln_b, tC, Cc);
  gemm(tC, t_m1w, m1.b, o_m1, Mrows, Cc, Cc, ACT_SIGMOID, stream);
  layernorm_k<<<Mrows, 256, 0, stream>>>(hC, m2.ln_g, m2.ln_b, tC, Cc);
  gemm(tC, t_m2w, m2.b, o_m2, Mrows, Cc, Cc, ACT_SIGMOID, stream);

  combine_k<<<nbC, 256, 0, stream>>>(z_mix, o_m1, o_r1, o_y1);
  combine_k<<<nbC, 256, 0, stream>>>(z_mix, o_m2, o_r2, o_y2);
}
